// LinearAttention_22548578304005
// MI455X (gfx1250) — matrix operand order repaired and checked by reading, not compiled, not run on hardware
//
#include <hip/hip_runtime.h>
#include <hip/hip_bf16.h>

// ---------------- problem constants ----------------
constexpr int Bn   = 4;
constexpr int Sn   = 4096;
constexpr int Hn   = 256;
constexpr int IN   = 512;     // INTER
constexpr int DEP  = 8;
constexpr int KT   = 7;
constexpr int OUTC = 256;
constexpr int PAD  = 16;      // zero rows in front of each (S,C) activation slab
constexpr float PI_F = 3.14159265358979323846f;

// ---------------- vector types ----------------
typedef __attribute__((ext_vector_type(16))) __bf16 bf16x16;
typedef __attribute__((ext_vector_type(8)))  __bf16 bf16x8;
typedef __attribute__((ext_vector_type(8)))  float  f32x8;

// ---------------- WMMA helper ----------------
__device__ inline f32x8 wmma_bf16(bf16x16 a, bf16x16 b, f32x8 c) {
  return __builtin_amdgcn_wmma_f32_16x16x32_bf16(
      /*neg_a=*/false, a, /*neg_b=*/false, b,
      /*c_mod=*/(short)0, c, /*reuse_a=*/false, /*reuse_b=*/false);
}

// A fragment loader: base already lane-adjusted (row r, +h*8 cols).
// Needs elements [0..7] and [16..23] relative to p.
__device__ inline bf16x16 load_a16(const __bf16* p) {
  bf16x8 lo = *reinterpret_cast<const bf16x8*>(p);
  bf16x8 hi = *reinterpret_cast<const bf16x8*>(p + 16);
  bf16x16 o;
#pragma unroll
  for (int i = 0; i < 8; ++i) { o[i] = lo[i]; o[i + 8] = hi[i]; }
  return o;
}

// ---------------------------------------------------------------------------
// Y(S x ND) = sum_t  X[s + t - (TAPS-1)] (S x KD)  *  Wt^T (KD x ND)
// X: bf16 (Bn, Sn+PAD, KD), data starts at row PAD, rows [0,PAD) are zero.
// W: bf16 (TAPS, ND, KD) row-major.  Y: f32 (Bn, Sn, ND).
// Wave tile 64x64: 4 A frags x 4 B frags -> 16 WMMAs per 16-load k-step (1:1
// load:wmma issue ratio).  Taps are row-contiguous, so the flattened (tap,k)
// iteration i addresses A at base + i*32.  Software-pipelined double buffer.
// ND is a template parameter so all store addresses fold into immediate
// offsets off a single base (keeps VGPR count under 256, no address chains).
// Block: 8 waves as 2(M) x 4(N) => 128 x 256 tile.
// ---------------------------------------------------------------------------
template <int TAPS, int KD, int ND>
__global__ __launch_bounds__(256) void gemm_bf16_taps(
    const __bf16* __restrict__ X, const __bf16* __restrict__ W,
    float* __restrict__ Y) {
  constexpr int KSTEPS = KD / 32;          // power of two (8 or 16)
  constexpr int TOTAL  = TAPS * KSTEPS;
  constexpr int KSH    = (KSTEPS == 8) ? 3 : 4;

  const int lane = threadIdx.x & 31;
  const int wave = threadIdx.x >> 5;
  const int wm = wave & 1;   // 2 waves along M (64 rows each)
  const int wn = wave >> 1;  // 4 waves along N (64 cols each)
  const int m0 = blockIdx.y * 128 + wm * 64;
  const int n0 = blockIdx.x * 256 + wn * 64;
  const int r = lane & 15, h = lane >> 4;

  const __bf16* Xb = X + (size_t)blockIdx.z * (Sn + PAD) * KD;
  float* Yb = Y + (size_t)blockIdx.z * Sn * ND;

  // Lane-adjusted base pointers.
  const __bf16* aBase = Xb + (size_t)(PAD + m0 + r - (TAPS - 1)) * KD + h * 8;
  const __bf16* wBase = W + (size_t)(n0 + r) * KD + h * 8;

  f32x8 acc[4][4];
#pragma unroll
  for (int i = 0; i < 4; ++i)
#pragma unroll
    for (int j = 0; j < 4; ++j)
#pragma unroll
      for (int e = 0; e < 8; ++e) acc[i][j][e] = 0.0f;

  bf16x16 ac[4], bc[4];

  // -------- prologue: load iteration 0 --------
#pragma unroll
  for (int f = 0; f < 4; ++f) ac[f] = load_a16(aBase + (size_t)(f * 16 * KD));
#pragma unroll
  for (int j = 0; j < 4; ++j)
    bc[j] = load_a16(wBase + (size_t)(j * 16 * KD));

  // -------- pipelined main loop --------
#pragma unroll 2
  for (int i = 1; i < TOTAL; ++i) {
    // issue loads for step i
    const __bf16* pa = aBase + (size_t)i * 32;   // taps are row-contiguous
    bf16x16 an[4];
#pragma unroll
    for (int f = 0; f < 4; ++f) an[f] = load_a16(pa + (size_t)(f * 16 * KD));
    const int t  = i >> KSH;
    const int k0 = (i & (KSTEPS - 1)) * 32;
    const __bf16* pw = wBase + (size_t)t * (ND * KD) + k0;
    bf16x16 bn[4];
#pragma unroll
    for (int j = 0; j < 4; ++j)
      bn[j] = load_a16(pw + (size_t)(j * 16 * KD));

    // compute step i-1 while step i loads are in flight
#pragma unroll
    for (int j = 0; j < 4; ++j)
#pragma unroll
      for (int f = 0; f < 4; ++f)
        acc[f][j] = wmma_bf16(ac[f], bc[j], acc[f][j]);
#pragma unroll
    for (int f = 0; f < 4; ++f) ac[f] = an[f];
#pragma unroll
    for (int j = 0; j < 4; ++j) bc[j] = bn[j];
  }

  // -------- epilogue: last step --------
#pragma unroll
  for (int j = 0; j < 4; ++j)
#pragma unroll
    for (int f = 0; f < 4; ++f)
      acc[f][j] = wmma_bf16(ac[f], bc[j], acc[f][j]);

  // -------- store: one base, all offsets compile-time immediates --------
  float* base = Yb + (size_t)(m0 + h * 8) * ND + (n0 + r);
#pragma unroll
  for (int i = 0; i < 4; ++i)
#pragma unroll
    for (int j = 0; j < 4; ++j)
#pragma unroll
      for (int e = 0; e < 8; ++e)
        base[i * 16 * ND + j * 16 + e * ND] = acc[i][j][e];
}

// ---------------- block reduction helpers (256 threads) ----------------
__device__ inline float blockSum(float v, float* sm) {
  int t = threadIdx.x;
  sm[t] = v; __syncthreads();
  for (int off = 128; off > 0; off >>= 1) {
    if (t < off) sm[t] += sm[t + off];
    __syncthreads();
  }
  float r = sm[0]; __syncthreads();
  return r;
}
__device__ inline float blockMax(float v, float* sm) {
  int t = threadIdx.x;
  sm[t] = v; __syncthreads();
  for (int off = 128; off > 0; off >>= 1) {
    if (t < off) sm[t] = fmaxf(sm[t], sm[t + off]);
    __syncthreads();
  }
  float r = sm[0]; __syncthreads();
  return r;
}

// ---------------- weight conversion ----------------
__global__ void cast_f32_bf16_kernel(const float* __restrict__ in,
                                     __bf16* __restrict__ out, long n) {
  for (long i = (long)blockIdx.x * blockDim.x + threadIdx.x; i < n;
       i += (long)gridDim.x * blockDim.x)
    out[i] = (__bf16)in[i];
}
// w1 (DEP,3,IN,IN,KT) f32 -> (DEP*3, KT, IN, IN) bf16 (tap-major for GEMM)
__global__ void convert_w1_kernel(const float* __restrict__ in,
                                  __bf16* __restrict__ out) {
  const long total = (long)DEP * 3 * KT * IN * IN;
  for (long i = (long)blockIdx.x * blockDim.x + threadIdx.x; i < total;
       i += (long)gridDim.x * blockDim.x) {
    long k = i % IN;
    long n = (i / IN) % IN;
    long t = (i / ((long)IN * IN)) % KT;
    long lb = i / ((long)IN * IN * KT);
    out[i] = (__bf16)in[((lb * IN + n) * IN + k) * KT + t];
  }
}

// ---------------- embedding gather ----------------
__global__ void embed_kernel(const float* __restrict__ emb,
                             const int* __restrict__ inp,
                             float* __restrict__ a, float* __restrict__ b) {
  int row = blockIdx.x;          // b*Sn + s
  int c = threadIdx.x;           // 0..255
  int tok = inp[row];
  a[(size_t)row * Hn + c] = emb[(size_t)tok * (2 * Hn) + c];
  b[(size_t)row * Hn + c] = emb[(size_t)tok * (2 * Hn) + Hn + c];
}

// ---------------- out = b + pos_embd, bf16, zero-padded rows ----------------
__global__ void addpos_kernel(const float* __restrict__ bstate,
                              __bf16* __restrict__ outp) {
  int rp = blockIdx.x;                   // over Bn*(Sn+PAD)
  int c = threadIdx.x;                   // 0..255
  int bb = rp / (Sn + PAD), sp = rp % (Sn + PAD);
  __bf16* dst = outp + (size_t)rp * Hn + c;
  if (sp < PAD) { *dst = (__bf16)0.0f; return; }
  int s = sp - PAD;
  float feat = c + 1.0f;
  float additive = fmodf(feat, 2.0f);
  float fe = (feat - additive) * 0.5f * (8.0f / Hn) - logf((float)Sn / (2.0f * PI_F));
  fe = expf(fe) + additive * PI_F;
  float pe = sinf((s + 1.0f) * fe) * (1.0f / DEP);
  *dst = (__bf16)(bstate[((size_t)bb * Sn + s) * Hn + c] + pe);
}

// ---------------- act_norm: f32 (B,S,N) -> bf16 (B,S+PAD,N) padded ----------------
__global__ void actnorm_kernel(const float* __restrict__ in,
                               __bf16* __restrict__ out, int N) {
  __shared__ float sm[256];
  int rp = blockIdx.x;
  int bb = rp / (Sn + PAD), sp = rp % (Sn + PAD);
  __bf16* dst = out + (size_t)rp * N;
  if (sp < PAD) {
    for (int i = threadIdx.x; i < N; i += 256) dst[i] = (__bf16)0.0f;
    return;
  }
  int s = sp - PAD;
  const float* x = in + ((size_t)bb * Sn + s) * N;
  const int V = N / 256;  // 1 or 2
  float v[2]; float lsum = 0.0f;
  for (int j = 0; j < V; ++j) {
    v[j] = fmaxf(x[threadIdx.x + j * 256], 0.0f);
    lsum += v[j];
  }
  float mean = blockSum(lsum, sm) / (float)N;
  float lss = 0.0f;
  for (int j = 0; j < V; ++j) { v[j] -= mean; lss += v[j] * v[j]; }
  float ss = blockSum(lss, sm);
  float scale = sqrtf((float)N) / (sqrtf(ss) + 1e-5f);
  for (int j = 0; j < V; ++j)
    dst[threadIdx.x + j * 256] = (__bf16)(v[j] * scale);
}

// ---------------- combine: y = act_norm(b*(cumsum_c(d)/pos + p) + s); a += y ----------------
__global__ void combine_kernel(const float* __restrict__ d,
                               const float* __restrict__ p,
                               const float* __restrict__ sb,
                               float* __restrict__ aBuf,
                               const float* __restrict__ bBuf) {
  __shared__ float sc[256];
  __shared__ float sm[256];
  int row = blockIdx.x;            // b*Sn + s
  int c = threadIdx.x;
  int s = row % Sn;
  size_t idx = (size_t)row * Hn + c;
  // inclusive scan over channels (axis=1 in NCH == channel dim here)
  sc[c] = d[idx]; __syncthreads();
  for (int off = 1; off < 256; off <<= 1) {
    float t = (c >= off) ? sc[c - off] : 0.0f;
    __syncthreads();
    sc[c] += t;
    __syncthreads();
  }
  float cd = sc[c];
  float y = bBuf[idx] * (cd / (s + 1.0f) + p[idx]) + sb[idx];
  float o = fmaxf(y, 0.0f);
  float mean = blockSum(o, sm) / 256.0f;
  o -= mean;
  float ss = blockSum(o * o, sm);
  y = o * (16.0f / (sqrtf(ss) + 1e-5f));   // sqrt(256)=16
  aBuf[idx] += y;                          // aBuf becomes the new b
}

// ---------------- concat [a;b] -> bf16 padded (B, S+PAD, 512) ----------------
__global__ void concat_kernel(const float* __restrict__ a,
                              const float* __restrict__ b,
                              __bf16* __restrict__ x2) {
  int rp = blockIdx.x;
  int c = threadIdx.x;
  int bb = rp / (Sn + PAD), sp = rp % (Sn + PAD);
  __bf16* dst = x2 + (size_t)rp * (2 * Hn);
  if (sp < PAD) { dst[c] = (__bf16)0.0f; dst[c + Hn] = (__bf16)0.0f; return; }
  size_t src = ((size_t)bb * Sn + (sp - PAD)) * Hn + c;
  dst[c] = (__bf16)a[src];
  dst[c + Hn] = (__bf16)b[src];
}

// ---------------- per-row NLL from logits ----------------
__global__ void nll_kernel(const float* __restrict__ logits,
                           const float* __restrict__ bias,
                           const int* __restrict__ tgt,
                           float* __restrict__ rownll) {
  __shared__ float sm[256];
  int row = blockIdx.x;
  int c = threadIdx.x;
  float z = logits[(size_t)row * OUTC + c] + bias[c];
  float mx = blockMax(z, sm);
  float sum = blockSum(expf(z - mx), sm);
  float lse = logf(sum) + mx;
  if (c == tgt[row]) rownll[row] = lse - z;
}

__global__ void reduce_kernel(const float* __restrict__ rownll,
                              float* __restrict__ out) {
  __shared__ float sm[256];
  float s = 0.0f;
  for (int i = threadIdx.x; i < Bn * Sn; i += 256) s += rownll[i];
  float total = blockSum(s, sm);
  if (threadIdx.x == 0) out[0] = total / (float)(Bn * Sn);
}

// ---------------- host driver ----------------
extern "C" void kernel_launch(void* const* d_in, const int* in_sizes, int n_in,
                              void* d_out, int out_size, void* d_ws,
                              size_t ws_size, hipStream_t stream) {
  (void)in_sizes; (void)n_in; (void)out_size; (void)ws_size;
  const float* emb   = (const float*)d_in[0];
  const float* w0f   = (const float*)d_in[1];
  const float* w1f   = (const float*)d_in[2];
  const float* w2f   = (const float*)d_in[3];
  const float* outwf = (const float*)d_in[4];
  const float* outb  = (const float*)d_in[5];
  const int*   inp   = (const int*)d_in[6];
  const int*   tgt   = (const int*)d_in[7];

  char* ws = (char*)d_ws;
  size_t off = 0;
  auto alloc = [&](size_t bytes) {
    size_t o = off;
    off += (bytes + 255) & ~(size_t)255;
    return o;
  };
  __bf16* w0h  = (__bf16*)(ws + alloc((size_t)DEP * 3 * IN * Hn * 2));
  __bf16* w1h  = (__bf16*)(ws + alloc((size_t)DEP * 3 * KT * IN * IN * 2));
  __bf16* w2h  = (__bf16*)(ws + alloc((size_t)DEP * 3 * Hn * IN * 2));
  __bf16* owh  = (__bf16*)(ws + alloc((size_t)OUTC * 2 * Hn * 2));
  float*  bufA = (float*)(ws + alloc((size_t)Bn * Sn * Hn * 4));
  float*  bufB = (float*)(ws + alloc((size_t)Bn * Sn * Hn * 4));
  __bf16* outp = (__bf16*)(ws + alloc((size_t)Bn * (Sn + PAD) * Hn * 2));
  float*  t1f  = (float*)(ws + alloc((size_t)Bn * Sn * IN * 4));
  __bf16* t1h  = (__bf16*)(ws + alloc((size_t)Bn * (Sn + PAD) * IN * 2));
  float*  dbuf = (float*)(ws + alloc((size_t)Bn * Sn * Hn * 4));
  float*  pbuf = (float*)(ws + alloc((size_t)Bn * Sn * Hn * 4));
  float*  sbuf = (float*)(ws + alloc((size_t)Bn * Sn * Hn * 4));
  __bf16* x2h  = (__bf16*)(ws + alloc((size_t)Bn * (Sn + PAD) * 2 * Hn * 2));
  float*  rnll = (float*)(ws + alloc((size_t)Bn * Sn * 4));
  float*  logits = t1f;  // reuse: free after the layer loop, size fits

  // --- weights -> bf16 (w1 re-laid tap-major) ---
  cast_f32_bf16_kernel<<<2048, 256, 0, stream>>>(w0f, w0h, (long)DEP * 3 * IN * Hn);
  convert_w1_kernel<<<4096, 256, 0, stream>>>(w1f, w1h);
  cast_f32_bf16_kernel<<<2048, 256, 0, stream>>>(w2f, w2h, (long)DEP * 3 * Hn * IN);
  cast_f32_bf16_kernel<<<512, 256, 0, stream>>>(outwf, owh, (long)OUTC * 2 * Hn);

  // --- embedding split ---
  embed_kernel<<<Bn * Sn, 256, 0, stream>>>(emb, inp, bufA, bufB);

  float* cura = bufA;  // 'a' carry
  float* curb = bufB;  // 'b' carry
  float* branch[3] = {dbuf, pbuf, sbuf};

  for (int l = 0; l < DEP; ++l) {
    addpos_kernel<<<Bn * (Sn + PAD), 256, 0, stream>>>(curb, outp);
    for (int br = 0; br < 3; ++br) {
      const __bf16* w0p = w0h + (size_t)(l * 3 + br) * IN * Hn;
      const __bf16* w1p = w1h + (size_t)(l * 3 + br) * KT * IN * IN;
      const __bf16* w2p = w2h + (size_t)(l * 3 + br) * Hn * IN;
      gemm_bf16_taps<1, Hn, IN>
          <<<dim3(IN / 256, Sn / 128, Bn), 256, 0, stream>>>(outp, w0p, t1f);
      actnorm_kernel<<<Bn * (Sn + PAD), 256, 0, stream>>>(t1f, t1h, IN);
      gemm_bf16_taps<KT, IN, IN>
          <<<dim3(IN / 256, Sn / 128, Bn), 256, 0, stream>>>(t1h, w1p, t1f);
      actnorm_kernel<<<Bn * (Sn + PAD), 256, 0, stream>>>(t1f, t1h, IN);
      gemm_bf16_taps<1, IN, Hn>
          <<<dim3(Hn / 256, Sn / 128, Bn), 256, 0, stream>>>(t1h, w2p, branch[br]);
    }
    combine_kernel<<<Bn * Sn, 256, 0, stream>>>(dbuf, pbuf, sbuf, cura, curb);
    float* tmp = cura; cura = curb; curb = tmp;  // (a,b) <- (b, a+y)
  }

  // --- head: logits = [a;b] @ out_w^T + bias, log_softmax, NLL mean ---
  concat_kernel<<<Bn * (Sn + PAD), 256, 0, stream>>>(cura, curb, x2h);
  gemm_bf16_taps<1, 2 * Hn, OUTC>
      <<<dim3(OUTC / 256, Sn / 128, Bn), 256, 0, stream>>>(x2h, owh, logits);
  nll_kernel<<<Bn * Sn, 256, 0, stream>>>(logits, outb, tgt, rnll);
  reduce_kernel<<<1, 256, 0, stream>>>(rnll, (float*)d_out);
}
